// MetaGraphSAGE_35158602285294
// MI455X (gfx1250) — matrix operand order repaired and checked by reading, not compiled, not run on hardware
//
#include <hip/hip_runtime.h>
#include <hip/hip_bf16.h>
#include <math.h>

#define META  3
#define NNODE 50000
#define NEDGE 640000
#define FDIM  128
#define HDIM  128
#define DDIM  64

typedef __attribute__((ext_vector_type(16))) __bf16 v16bf;
typedef __attribute__((ext_vector_type(8)))  __bf16 v8bf;
typedef __attribute__((ext_vector_type(4)))  __bf16 v4bf;
typedef __attribute__((ext_vector_type(8)))  float  v8f;

// ---------------- utility kernels ----------------

__global__ void zero_f32(float* p, long long n) {
  long long i = (long long)blockIdx.x * blockDim.x + threadIdx.x;
  long long stride = (long long)gridDim.x * blockDim.x;
  for (; i < n; i += stride) p[i] = 0.0f;
}

__global__ void cvt_bf16(const float* in, __bf16* out, long long n) {
  long long i = (long long)blockIdx.x * blockDim.x + threadIdx.x;
  long long stride = (long long)gridDim.x * blockDim.x;
  for (; i < n; i += stride) out[i] = (__bf16)in[i];
}

// degree: cnt[meta][dst] += 1
__global__ void degree_kernel(const int* __restrict__ ei, float* __restrict__ cnt) {
  int meta = blockIdx.z;
  int e = blockIdx.x * blockDim.x + threadIdx.x;
  if (e < NEDGE) {
    int dst = ei[(size_t)meta * 2 * NEDGE + NEDGE + e];
    atomicAdd(&cnt[(size_t)meta * NNODE + dst], 1.0f);
  }
}

// scatter-add fp32 features: one wave per edge, 4 channels per lane (C=128)
__global__ void scatter_f32(const int* __restrict__ ei, const float* __restrict__ feat,
                            float* __restrict__ agg, int C) {
  int meta = blockIdx.z;
  int lane = threadIdx.x & 31;
  int e = blockIdx.x * 8 + (threadIdx.x >> 5);
  const int* eim = ei + (size_t)meta * 2 * NEDGE;
  int src = eim[e];
  int dst = eim[NEDGE + e];
  const float4 v = *(const float4*)(feat + ((size_t)meta * NNODE + src) * C + lane * 4);
  float* dp = agg + ((size_t)meta * NNODE + dst) * C + lane * 4;
  atomicAdd(dp + 0, v.x);
  atomicAdd(dp + 1, v.y);
  atomicAdd(dp + 2, v.z);
  atomicAdd(dp + 3, v.w);
}

// scatter-add bf16 features (layer-2 input h), accumulate in fp32
__global__ void scatter_bf16(const int* __restrict__ ei, const __bf16* __restrict__ feat,
                             float* __restrict__ agg, int C) {
  int meta = blockIdx.z;
  int lane = threadIdx.x & 31;
  int e = blockIdx.x * 8 + (threadIdx.x >> 5);
  const int* eim = ei + (size_t)meta * 2 * NEDGE;
  int src = eim[e];
  int dst = eim[NEDGE + e];
  const v4bf v = *(const v4bf*)(feat + ((size_t)meta * NNODE + src) * C + lane * 4);
  float* dp = agg + ((size_t)meta * NNODE + dst) * C + lane * 4;
  atomicAdd(dp + 0, (float)v[0]);
  atomicAdd(dp + 1, (float)v[1]);
  atomicAdd(dp + 2, (float)v[2]);
  atomicAdd(dp + 3, (float)v[3]);
}

// mean = agg / max(cnt,1), convert to bf16
__global__ void mean_to_bf16(const float* __restrict__ agg, const float* __restrict__ cnt,
                             __bf16* __restrict__ out, long long n, int C) {
  long long i = (long long)blockIdx.x * blockDim.x + threadIdx.x;
  long long stride = (long long)gridDim.x * blockDim.x;
  for (; i < n; i += stride) {
    float c = cnt[i / C];
    c = c < 1.0f ? 1.0f : c;
    out[i] = (__bf16)(agg[i] / c);
  }
}

// ---------------- fused dual-GEMM with WMMA ----------------
// out = act( Aa @ Ba + Ab @ Bb + bias ) ; A: [M x 128] bf16 row-major,
// B: [128 x NCOLS] bf16 row-major. Block: 8 waves = 32 rows x 64 cols tile.
template <int NCOLS, bool ELU_BF16_OUT>
__global__ __launch_bounds__(256) void sage_gemm(
    const __bf16* __restrict__ Aa, const __bf16* __restrict__ Ab,
    const __bf16* __restrict__ Ba, const __bf16* __restrict__ Bb,
    const float* __restrict__ bias, __bf16* __restrict__ out_bf,
    float* __restrict__ out_f32, int M) {
  const int K = 128;
  int meta = blockIdx.z;
  Aa += (size_t)meta * M * K;
  Ab += (size_t)meta * M * K;
  Ba += (size_t)meta * K * NCOLS;
  Bb += (size_t)meta * K * NCOLS;
  bias += (size_t)meta * NCOLS;

  int lane = threadIdx.x & 31;
  int w = threadIdx.x >> 5;
  int rowbase = blockIdx.x * 32 + (w >> 2) * 16;      // 2 row-tiles
  int nbase   = blockIdx.y * 64 + (w & 3) * 16;       // 4 col-tiles

  // A layout: lane l -> row m = l&15, K-offset group kb = (l>>4)*8
  int m = rowbase + (lane & 15);
  int mc = m < M ? m : M - 1;                          // clamp reads; stores predicated
  int kb = (lane >> 4) * 8;
  const __bf16* arA = Aa + (size_t)mc * K;
  const __bf16* arB = Ab + (size_t)mc * K;

  v8f acc = {};
#pragma unroll
  for (int koff = 0; koff < K; koff += 32) {
    v16bf aA, aB;
    {
      v8bf lo = *(const v8bf*)(arA + koff + kb);
      v8bf hi = *(const v8bf*)(arA + koff + 16 + kb);
#pragma unroll
      for (int i = 0; i < 8; ++i) { aA[i] = lo[i]; aA[i + 8] = hi[i]; }
    }
    {
      v8bf lo = *(const v8bf*)(arB + koff + kb);
      v8bf hi = *(const v8bf*)(arB + koff + 16 + kb);
#pragma unroll
      for (int i = 0; i < 8; ++i) { aB[i] = lo[i]; aB[i + 8] = hi[i]; }
    }
    // B layout: lane l -> column n = nbase + (l&15), k in same order as A
    v16bf bA, bB;
#pragma unroll
    for (int i = 0; i < 8; ++i) {
      bA[i]     = Ba[(size_t)(koff + kb + i) * NCOLS + nbase + (lane & 15)];
      bA[i + 8] = Ba[(size_t)(koff + 16 + kb + i) * NCOLS + nbase + (lane & 15)];
      bB[i]     = Bb[(size_t)(koff + kb + i) * NCOLS + nbase + (lane & 15)];
      bB[i + 8] = Bb[(size_t)(koff + 16 + kb + i) * NCOLS + nbase + (lane & 15)];
    }
    acc = __builtin_amdgcn_wmma_f32_16x16x32_bf16(false, aA, false, bA,
                                                  (short)0, acc, false, false);
    acc = __builtin_amdgcn_wmma_f32_16x16x32_bf16(false, aB, false, bB,
                                                  (short)0, acc, false, false);
  }

  int col = nbase + (lane & 15);
  float bv = bias[col];
  size_t obase = (size_t)meta * M * NCOLS;
#pragma unroll
  for (int r = 0; r < 8; ++r) {
    int row = rowbase + (lane >> 4) * 8 + r;          // C/D layout
    if (row < M) {
      float v = acc[r] + bv;
      if (ELU_BF16_OUT) {
        v = v > 0.0f ? v : (__expf(v) - 1.0f);        // ELU (alpha=1)
        out_bf[obase + (size_t)row * NCOLS + col] = (__bf16)v;
      } else {
        out_f32[obase + (size_t)row * NCOLS + col] = v;
      }
    }
  }
}

// ---------------- log-softmax over D=64, one wave per row ----------------
__global__ void log_softmax64(float* __restrict__ out, int rows) {
  int w = threadIdx.x >> 5;
  int lane = threadIdx.x & 31;
  int r = blockIdx.x * 8 + w;
  if (r >= rows) return;
  float* p = out + (size_t)r * 64;
  float v0 = p[lane];
  float v1 = p[lane + 32];
  float mx = fmaxf(v0, v1);
#pragma unroll
  for (int off = 16; off > 0; off >>= 1) mx = fmaxf(mx, __shfl_xor(mx, off, 32));
  float s = __expf(v0 - mx) + __expf(v1 - mx);
#pragma unroll
  for (int off = 16; off > 0; off >>= 1) s += __shfl_xor(s, off, 32);
  float l = mx + __logf(s);
  p[lane] = v0 - l;
  p[lane + 32] = v1 - l;
}

// ---------------- driver ----------------
extern "C" void kernel_launch(void* const* d_in, const int* in_sizes, int n_in,
                              void* d_out, int out_size, void* d_ws, size_t ws_size,
                              hipStream_t stream) {
  const float* x   = (const float*)d_in[0];
  const int*   ei  = (const int*)d_in[1];
  const float* W1l = (const float*)d_in[2];
  const float* W1r = (const float*)d_in[3];
  const float* b1  = (const float*)d_in[4];
  const float* W2l = (const float*)d_in[5];
  const float* W2r = (const float*)d_in[6];
  const float* b2  = (const float*)d_in[7];

  char* ws = (char*)d_ws;
  size_t off = 0;
  auto alloc = [&](size_t bytes) -> void* {
    off = (off + 255) & ~(size_t)255;
    void* p = ws + off;
    off += bytes;
    return p;
  };

  float*  cnt     = (float*)alloc(sizeof(float) * (size_t)META * NNODE);
  float*  agg     = (float*)alloc(sizeof(float) * (size_t)META * NNODE * FDIM);
  __bf16* x_bf    = (__bf16*)alloc(2ull * META * NNODE * FDIM);
  __bf16* mean_bf = (__bf16*)alloc(2ull * META * NNODE * FDIM);
  __bf16* h_bf    = (__bf16*)alloc(2ull * META * NNODE * HDIM);
  __bf16* W1l_bf  = (__bf16*)alloc(2ull * META * FDIM * HDIM);
  __bf16* W1r_bf  = (__bf16*)alloc(2ull * META * FDIM * HDIM);
  __bf16* W2l_bf  = (__bf16*)alloc(2ull * META * HDIM * DDIM);
  __bf16* W2r_bf  = (__bf16*)alloc(2ull * META * HDIM * DDIM);

  dim3 b256(256);
  const long long nfeat = (long long)META * NNODE * FDIM;

  // layer-1 aggregation
  zero_f32<<<dim3(1024), b256, 0, stream>>>(cnt, (long long)META * NNODE);
  zero_f32<<<dim3(8192), b256, 0, stream>>>(agg, nfeat);
  degree_kernel<<<dim3((NEDGE + 255) / 256, 1, META), b256, 0, stream>>>(ei, cnt);
  cvt_bf16<<<dim3(8192), b256, 0, stream>>>(x, x_bf, nfeat);
  cvt_bf16<<<dim3(64), b256, 0, stream>>>(W1l, W1l_bf, (long long)META * FDIM * HDIM);
  cvt_bf16<<<dim3(64), b256, 0, stream>>>(W1r, W1r_bf, (long long)META * FDIM * HDIM);
  cvt_bf16<<<dim3(32), b256, 0, stream>>>(W2l, W2l_bf, (long long)META * HDIM * DDIM);
  cvt_bf16<<<dim3(32), b256, 0, stream>>>(W2r, W2r_bf, (long long)META * HDIM * DDIM);
  scatter_f32<<<dim3(NEDGE / 8, 1, META), b256, 0, stream>>>(ei, x, agg, FDIM);
  mean_to_bf16<<<dim3(8192), b256, 0, stream>>>(agg, cnt, mean_bf, nfeat, FDIM);

  // layer 1: h = ELU(mean@W1l + x@W1r + b1), stored bf16
  sage_gemm<HDIM, true><<<dim3((NNODE + 31) / 32, HDIM / 64, META), b256, 0, stream>>>(
      mean_bf, x_bf, W1l_bf, W1r_bf, b1, h_bf, nullptr, NNODE);

  // layer-2 aggregation on h
  zero_f32<<<dim3(8192), b256, 0, stream>>>(agg, (long long)META * NNODE * HDIM);
  scatter_bf16<<<dim3(NEDGE / 8, 1, META), b256, 0, stream>>>(ei, h_bf, agg, HDIM);
  mean_to_bf16<<<dim3(8192), b256, 0, stream>>>(agg, cnt, mean_bf,
                                                (long long)META * NNODE * HDIM, HDIM);

  // layer 2: out = mean@W2l + h@W2r + b2 (fp32 into d_out)
  sage_gemm<DDIM, false><<<dim3((NNODE + 31) / 32, 1, META), b256, 0, stream>>>(
      mean_bf, h_bf, W2l_bf, W2r_bf, b2, nullptr, (float*)d_out, NNODE);

  // log-softmax in place
  log_softmax64<<<dim3((META * NNODE) / 8), b256, 0, stream>>>((float*)d_out, META * NNODE);
}
